// DeformConvTorchvision_13417477833365
// MI455X (gfx1250) — matrix operand order repaired and checked by reading, not compiled, not run on hardware
//
#include <hip/hip_runtime.h>
#include <math.h>

// ---------------- problem constants (from reference setup_inputs) ------------
constexpr int Bn   = 4;            // batch
constexpr int C    = 256;          // channels
constexpr int H    = 96, W = 96;
constexpr int HW   = H * W;        // 9216
constexpr int G    = 4;            // groups
constexpr int Cg   = C / G;        // 64
constexpr int K    = 9;            // 3x3 taps
constexpr int KRED = Cg * K;       // 576 reduction depth per group
constexpr int OFFC = 2 * K;        // 18 offset channels

constexpr int NPIX   = 32;         // pixels per block tile
constexpr int NCHUNK = KRED / 32;  // 18 k-chunks of 32
constexpr int NTILES = HW / NPIX;  // 288 pixel tiles
constexpr int BSTR   = 40;         // padded LDS stride (ushorts) for B staging

// ---------------- types for WMMA fragments -----------------------------------
typedef __attribute__((ext_vector_type(16))) __bf16        bf16x16;
typedef __attribute__((ext_vector_type(8)))  float         f32x8;
typedef __attribute__((ext_vector_type(4)))  unsigned int  u32x4;

union Frag {
    bf16x16 bf;
    u32x4   u4[2];
};

struct SampEntry {      // bilinear descriptor for one (pixel, tap)
    float w[4];         // per-corner weights, zeroed when out of bounds
    int   o[4];         // clamped y*W+x gather offsets
};

__device__ __forceinline__ unsigned short f32_to_bf16(float f) {
    unsigned int u = __builtin_bit_cast(unsigned int, f);
    u += 0x7FFFu + ((u >> 16) & 1u);      // round-to-nearest-even
    return (unsigned short)(u >> 16);
}

// ---------------- kernel 1: weight fp32 -> bf16 (layout already GEMM-A) ------
__global__ void k_w2bf(const float* __restrict__ w, unsigned short* __restrict__ o, int n) {
    int t = blockIdx.x * 256 + threadIdx.x;
    if (t < n) o[t] = f32_to_bf16(w[t]);
}

// ---------------- kernel 2: offset conv (fp32 direct, 18 outs/thread) --------
__global__ __launch_bounds__(256) void k_offset_conv(const float* __restrict__ x,
                                                     const float* __restrict__ ow,
                                                     const float* __restrict__ ob,
                                                     float* __restrict__ offs) {
    int t = blockIdx.x * 256 + threadIdx.x;    // (b, pixel)
    if (t >= Bn * HW) return;
    const int b  = t / HW;
    const int pg = t - b * HW;
    const int y  = pg / W;
    const int xx = pg - y * W;

    float acc[OFFC];
#pragma unroll
    for (int o = 0; o < OFFC; ++o) acc[o] = ob[o];

    for (int c = 0; c < C; ++c) {
        const float* xb = x + (size_t)(b * C + c) * HW;
        float v[K];
#pragma unroll
        for (int kh = 0; kh < 3; ++kh)
#pragma unroll
            for (int kw = 0; kw < 3; ++kw) {
                const int yy = y + kh - 1, xc = xx + kw - 1;
                const bool ok = (yy >= 0) && (yy < H) && (xc >= 0) && (xc < W);
                v[kh * 3 + kw] = ok ? xb[yy * W + xc] : 0.0f;
            }
#pragma unroll
        for (int o = 0; o < OFFC; ++o) {
            const float* wr = ow + ((size_t)o * C + c) * K;  // uniform -> scalar loads
#pragma unroll
            for (int tap = 0; tap < K; ++tap)
                acc[o] = fmaf(v[tap], wr[tap], acc[o]);
        }
    }
#pragma unroll
    for (int o = 0; o < OFFC; ++o)
        offs[((size_t)(b * OFFC + o) * H + y) * W + xx] = acc[o];
}

// ---------------- kernel 3: fused bilinear sampling + grouped WMMA GEMM ------
__global__ __launch_bounds__(256) void k_deform(const float* __restrict__ x,
                                                const float* __restrict__ offs,
                                                const unsigned short* __restrict__ wbf,
                                                float* __restrict__ out) {
    __shared__ SampEntry s_samp[NPIX * K];                     // 288 * 32B = 9216B
    __shared__ __align__(16) unsigned short s_B[NPIX * BSTR];  // 2560B, padded stride

    const int tid   = threadIdx.x;
    const int lane  = tid & 31;
    const int wave  = tid >> 5;
    const int mtile = wave >> 1;      // 0..3 : output rows mtile*16..+15
    const int ntile = wave & 1;       // 0..1 : pixels ntile*16..+15 in tile
    const int half  = lane >> 4;
    const int l16   = lane & 15;

    int bid = blockIdx.x;
    const int tile = bid % NTILES;  bid /= NTILES;
    const int g    = bid % G;
    const int b    = bid / G;
    const int pix0 = tile * NPIX;

    // ---- phase 1: bilinear descriptors for 32 pixels x 9 taps ----
    for (int e = tid; e < NPIX * K; e += 256) {
        const int p  = e / K;
        const int kk = e - p * K;
        const int pg = pix0 + p;
        const int y  = pg / W;
        const int xx = pg - y * W;
        const float dy = offs[((size_t)(b * OFFC + 2 * kk)     * H + y) * W + xx];
        const float dx = offs[((size_t)(b * OFFC + 2 * kk + 1) * H + y) * W + xx];
        const int ky = kk / 3, kx = kk - ky * 3;
        const float py = (float)(y + ky - 1) + dy;
        const float px = (float)(xx + kx - 1) + dx;
        const bool valid = (py > -1.0f) && (py < (float)H) && (px > -1.0f) && (px < (float)W);
        const float y0f = floorf(py), x0f = floorf(px);
        const float ly = py - y0f, lx = px - x0f;
        const int y0 = (int)y0f, x0 = (int)x0f;
        const float wy[2]  = {1.0f - ly, ly};
        const float wxv[2] = {1.0f - lx, lx};
        SampEntry se;
#pragma unroll
        for (int j = 0; j < 4; ++j) {
            const int yi = y0 + (j >> 1), xi = x0 + (j & 1);
            const bool ok = valid && (yi >= 0) && (yi < H) && (xi >= 0) && (xi < W);
            const int yc = yi < 0 ? 0 : (yi > H - 1 ? H - 1 : yi);
            const int xc = xi < 0 ? 0 : (xi > W - 1 ? W - 1 : xi);
            se.w[j] = ok ? wy[j >> 1] * wxv[j & 1] : 0.0f;
            se.o[j] = yc * W + xc;
        }
        s_samp[e] = se;
    }
    __syncthreads();

    // ---- phase 2: k-chunk loop: sample->LDS, then v_wmma_f32_16x16x32_bf16 --
    f32x8 acc = {};
    const unsigned short* Ag = wbf + (size_t)g * Cg * KRED;   // [64][576] bf16
    const int arow = mtile * 16 + l16;

    for (int chunk = 0; chunk < NCHUNK; ++chunk) {
        // fill 32(red) x 32(pixel) bf16 tile: 4 values per thread
#pragma unroll
        for (int i = 0; i < 4; ++i) {
            const int idx = tid + i * 256;
            const int p   = idx & 31;        // lane == pixel -> coalesced gathers
            const int r   = idx >> 5;        // red within chunk
            const int red = chunk * 32 + r;
            const int c   = red / 9;
            const int kk  = red - c * 9;
            const SampEntry se = s_samp[p * K + kk];
            const float* xb = x + (size_t)(b * C + g * Cg + c) * HW;
            const float v = se.w[0] * xb[se.o[0]] + se.w[1] * xb[se.o[1]] +
                            se.w[2] * xb[se.o[2]] + se.w[3] * xb[se.o[3]];
            s_B[p * BSTR + r] = f32_to_bf16(v);
        }
        __syncthreads();

        // A fragment from global (L2-resident weights), ISA 16-bit 16x32 layout
        Frag afrag, bfrag;
        {
            const unsigned short* ap = Ag + (size_t)arow * KRED + chunk * 32;
            afrag.u4[0] = *reinterpret_cast<const u32x4*>(ap + half * 8);       // K 0..7 / 8..15
            afrag.u4[1] = *reinterpret_cast<const u32x4*>(ap + 16 + half * 8);  // K 16..23 / 24..31
            // B fragment from LDS: column = pixel, K contiguous per column
            const unsigned short* bp = &s_B[(ntile * 16 + l16) * BSTR + half * 8];
            bfrag.u4[0] = *reinterpret_cast<const u32x4*>(bp);
            bfrag.u4[1] = *reinterpret_cast<const u32x4*>(bp + 16);
        }
        acc = __builtin_amdgcn_wmma_f32_16x16x32_bf16(false, afrag.bf,
                                                      false, bfrag.bf,
                                                      (short)0, acc,
                                                      false, false);
        __syncthreads();
    }

    // ---- store D: VGPR r -> row (mtile*16 + r + 8*half), col ntile*16 + l16 -
    float* ob = out + (size_t)(b * C + g * Cg + mtile * 16) * HW + pix0 + ntile * 16 + l16;
#pragma unroll
    for (int r = 0; r < 8; ++r)
        ob[(size_t)(r + 8 * half) * HW] = acc[r];
}

// ---------------- launcher ---------------------------------------------------
extern "C" void kernel_launch(void* const* d_in, const int* in_sizes, int n_in,
                              void* d_out, int out_size, void* d_ws, size_t ws_size,
                              hipStream_t stream) {
    (void)in_sizes; (void)n_in; (void)out_size; (void)ws_size;
    const float* x        = (const float*)d_in[0];
    const float* offset_w = (const float*)d_in[1];
    const float* offset_b = (const float*)d_in[2];
    const float* weight   = (const float*)d_in[3];
    float*       out      = (float*)d_out;

    // workspace: [offsets fp32 (2,654,208 B)][weights bf16 (294,912 B)]
    float* offs = (float*)d_ws;
    const size_t offBytes = (size_t)Bn * OFFC * HW * sizeof(float);   // 256-aligned
    unsigned short* wbf = (unsigned short*)((char*)d_ws + offBytes);

    const int nW = C * Cg * K;                               // 147456
    k_w2bf<<<(nW + 255) / 256, 256, 0, stream>>>(weight, wbf, nW);

    const int nPix = Bn * HW;                                // 36864
    k_offset_conv<<<(nPix + 255) / 256, 256, 0, stream>>>(x, offset_w, offset_b, offs);

    k_deform<<<Bn * G * NTILES, 256, 0, stream>>>(x, offs, wbf, out);
}
